// SceneConstructionModel_10024453668877
// MI455X (gfx1250) — matrix operand order repaired and checked by reading, not compiled, not run on hardware
//
#include <hip/hip_runtime.h>
#include <hip/hip_bf16.h>
#include <math.h>

// ---------------------------------------------------------------------------
// Edge-featured GAT x2 + factorized link decoder for MI455X (gfx1250).
// Dense GEMMs: bf16 pre-converted operands, async global->LDS double-buffered
// pipeline, register-blocked v_wmma_f32_16x16x32_bf16 (4 per K-step per wave).
// ---------------------------------------------------------------------------

#define N_NODES   20000
#define E_EDGES   160000
#define F_IN      64
#define EDGE_DIM  8
#define H_HEADS   4
#define C_CH      128
#define HC        512            // H*C
#define EP        (E_EDGES + N_NODES)   // edges incl. self loops
#define DEC_H     256
#define NUM_RELS  4
#define M_PAD     20096          // ceil(20000/128)*128

typedef __attribute__((ext_vector_type(16))) __bf16 v16bf;
typedef __attribute__((ext_vector_type(8)))  float  v8f;
typedef int v4i_vs __attribute__((vector_size(16)));   // matches builtin param

#if __has_builtin(__builtin_amdgcn_global_load_async_to_lds_b128)
#define HAS_ASYNC_LDS 1
#else
#define HAS_ASYNC_LDS 0
#endif

#if HAS_ASYNC_LDS
#if __has_builtin(__builtin_amdgcn_s_wait_asynccnt)
#define WAIT_ASYNC(n) __builtin_amdgcn_s_wait_asynccnt(n)
#else
#define WAIT_ASYNC(n) asm volatile("s_wait_asynccnt %0" ::"i"(n) : "memory")
#endif
#else
#define WAIT_ASYNC(n) ((void)0)
#endif

// ------------------------------ utility ------------------------------------

__global__ void k_fill(float* __restrict__ p, float v, int n) {
    int i = blockIdx.x * blockDim.x + threadIdx.x;
    if (i < n) p[i] = v;
}

__device__ __forceinline__ void atomicMaxF(float* addr, float val) {
    if (val >= 0.0f)
        atomicMax((int*)addr, __float_as_int(val));
    else
        atomicMin((unsigned int*)addr, __float_as_uint(val));
}

__device__ __forceinline__ unsigned pack_bf16(float x, float y) {
    __bf16 a = (__bf16)x, b = (__bf16)y;
    unsigned short ua = __builtin_bit_cast(unsigned short, a);
    unsigned short ub = __builtin_bit_cast(unsigned short, b);
    return (unsigned)ua | ((unsigned)ub << 16);
}

// copy 16 bytes global -> LDS (async on CDNA5, sync fallback otherwise)
typedef v4i_vs __attribute__((address_space(1))) AS1v4i;
typedef v4i_vs __attribute__((address_space(3))) AS3v4i;

__device__ __forceinline__ void copy16_g2l(const void* g, void* l) {
#if HAS_ASYNC_LDS
    __builtin_amdgcn_global_load_async_to_lds_b128(
        (AS1v4i*)(void*)g, (AS3v4i*)l, 0, 0);
#else
    *(uint4*)l = *(const uint4*)g;
#endif
}

// ---------------------- fp32 -> bf16 conversions ---------------------------

// rows x cols fp32 -> rowsPad x cols bf16 (packed pairs), pad rows zeroed.
__global__ void k_cvt_bf16_pad(const float* __restrict__ in,
                               unsigned* __restrict__ out,
                               int rows, int cols, int rowsPad) {
    int idx = blockIdx.x * blockDim.x + threadIdx.x;   // u32 pair index
    int total = (rowsPad * cols) >> 1;
    if (idx >= total) return;
    int j = idx << 1;
    int row = j / cols;
    out[idx] = (row < rows) ? pack_bf16(in[j], in[j + 1]) : 0u;
}

// B [K,Nc] fp32 row-major -> BT [Nc,K] bf16 row-major
__global__ void k_cvt_transpose_bf16(const float* __restrict__ B,
                                     unsigned short* __restrict__ BT,
                                     int K, int Nc) {
    int o = blockIdx.x * blockDim.x + threadIdx.x;     // o = n*K + k
    if (o >= K * Nc) return;
    int n = o / K, k = o - n * K;
    __bf16 b = (__bf16)B[(size_t)k * Nc + n];
    BT[o] = __builtin_bit_cast(unsigned short, b);
}

// ------------------- self-loop edge-attr mean (fill_value='mean') ----------

__global__ void k_edge_mean_accum(const int* __restrict__ ei,
                                  const float* __restrict__ ef,
                                  float* __restrict__ sum,
                                  float* __restrict__ cnt) {
    int e = blockIdx.x * blockDim.x + threadIdx.x;
    if (e >= E_EDGES) return;
    int d = ei[E_EDGES + e];
#pragma unroll
    for (int k = 0; k < EDGE_DIM; ++k)
        atomicAdd(&sum[d * EDGE_DIM + k], ef[e * EDGE_DIM + k]);
    atomicAdd(&cnt[d], 1.0f);
}

__global__ void k_edge_mean_div(float* __restrict__ sum,
                                const float* __restrict__ cnt) {
    int n = blockIdx.x * blockDim.x + threadIdx.x;
    if (n >= N_NODES) return;
    float c = fmaxf(cnt[n], 1.0f);
#pragma unroll
    for (int k = 0; k < EDGE_DIM; ++k)
        sum[n * EDGE_DIM + k] /= c;
}

// ------------------------------ bf16 WMMA GEMM -----------------------------
// C[M,Nc] = A[Mpad,K](bf16) @ BT[Nc,K](bf16)^T ; fp32 out.
// K%32==0, Nc%64==0. Grid (Mpad/128, Nc/64), 256 threads (8 waves).
// Double-buffered async global->LDS tile pipeline; 4 WMMAs per wave per step.

__global__ void __launch_bounds__(256)
k_gemm_bf16_wmma(const unsigned short* __restrict__ A,
                 const unsigned short* __restrict__ BT,
                 float* __restrict__ C,
                 int M, int K, int Nc) {
    // double-buffered bf16 tiles: A 128x32 (8KB each), B 64x32 (4KB each)
    __shared__ __align__(16) unsigned lA32[2 * 128 * 16];
    __shared__ __align__(16) unsigned lB32[2 * 64 * 16];

    const int t       = threadIdx.x;
    const int wave    = t >> 5;
    const int lane    = t & 31;
    const int lane_lo = lane & 15;
    const int hi      = lane >> 4;
    const int wm      = wave & 3;        // M group
    const int wn      = wave >> 2;       // N group

    const int tileM = blockIdx.x * 128;
    const int tileN = blockIdx.y * 64;

    const char* gA = (const char*)A;
    const char* gB = (const char*)BT;
    char*       sA = (char*)lA32;
    char*       sB = (char*)lB32;

    // issue one 32-K tile copy: 3 x 16B per thread
#define ISSUE_TILE(k0, buf)                                                     \
    do {                                                                        \
        _Pragma("unroll")                                                       \
        for (int i_ = 0; i_ < 2; ++i_) {                                        \
            int c_ = t + i_ * 256;          /* 0..511 */                        \
            int row_ = c_ >> 2, q_ = c_ & 3;                                    \
            copy16_g2l(gA + ((size_t)(tileM + row_) * K + (k0)) * 2 + q_ * 16,  \
                       sA + (buf) * 8192 + row_ * 64 + q_ * 16);                \
        }                                                                       \
        {                                                                       \
            int n_ = t >> 2, q_ = t & 3;                                        \
            copy16_g2l(gB + ((size_t)(tileN + n_) * K + (k0)) * 2 + q_ * 16,    \
                       sB + (buf) * 4096 + n_ * 64 + q_ * 16);                  \
        }                                                                       \
    } while (0)

    v8f acc[2][2] = {};
    const int T = K >> 5;

    ISSUE_TILE(0, 0);

    for (int i = 0; i < T; ++i) {
        if (i + 1 < T) {
            __syncthreads();                 // prior compute on bufNext done
            ISSUE_TILE((i + 1) << 5, (i + 1) & 1);
            WAIT_ASYNC(3);                   // tile i (in-order) has landed
        } else {
            WAIT_ASYNC(0);
        }
        __syncthreads();                     // everyone's tile-i copies visible

        const unsigned* a0 = lA32 + (i & 1) * (128 * 16);
        const unsigned* b0 = lB32 + (i & 1) * (64 * 16);

        union { v16bf v; unsigned u[8]; } af[2], bfr[2];
#pragma unroll
        for (int mi = 0; mi < 2; ++mi) {
            int m = wm * 32 + mi * 16 + lane_lo;
#pragma unroll
            for (int r = 0; r < 8; ++r) {
                int pair = r + hi * 8;
                af[mi].u[r] = a0[m * 16 + pair];
            }
        }
#pragma unroll
        for (int ni = 0; ni < 2; ++ni) {
            int n = wn * 32 + ni * 16 + lane_lo;
#pragma unroll
            for (int r = 0; r < 8; ++r)
                bfr[ni].u[r] = b0[n * 16 + r + hi * 8];
        }
#pragma unroll
        for (int mi = 0; mi < 2; ++mi)
#pragma unroll
            for (int ni = 0; ni < 2; ++ni)
                acc[mi][ni] = __builtin_amdgcn_wmma_f32_16x16x32_bf16(
                    false, af[mi].v, false, bfr[ni].v, (short)0, acc[mi][ni],
                    false, false);
    }
#undef ISSUE_TILE

    // ---- write C (elem r -> row +r+8*hi, col lane_lo) ----
#pragma unroll
    for (int mi = 0; mi < 2; ++mi) {
#pragma unroll
        for (int ni = 0; ni < 2; ++ni) {
#pragma unroll
            for (int r = 0; r < 8; ++r) {
                int gRow = tileM + wm * 32 + mi * 16 + r + 8 * hi;
                int gCol = tileN + wn * 32 + ni * 16 + lane_lo;
                if (gRow < M)
                    C[(size_t)gRow * Nc + gCol] = acc[mi][ni][r];
            }
        }
    }
}

// -------------------- per-node attention logits (alpha_s/alpha_d) ----------

__global__ void k_node_alpha(const float* __restrict__ xp,
                             const float* __restrict__ a_s,
                             const float* __restrict__ a_d,
                             float* __restrict__ as_,
                             float* __restrict__ ad_) {
    int t = blockIdx.x * blockDim.x + threadIdx.x;
    if (t >= N_NODES * H_HEADS) return;
    int n = t >> 2, h = t & 3;
    const float* xr = xp + (size_t)n * HC + h * C_CH;
    const float* vs = a_s + h * C_CH;
    const float* vd = a_d + h * C_CH;
    float ss = 0.f, sd = 0.f;
    for (int c = 0; c < C_CH; ++c) {
        float x = xr[c];
        ss += x * vs[c];
        sd += x * vd[c];
    }
    as_[t] = ss;
    ad_[t] = sd;
}

// ------------------ M_e[d,h] = sum_c We[d, h*C+c] * a_e[h,c] ---------------

__global__ void k_reduce_We_ae(const float* __restrict__ We,
                               const float* __restrict__ a_e,
                               float* __restrict__ Me) {
    int t = threadIdx.x;   // 32 threads: (d,h)
    if (t >= EDGE_DIM * H_HEADS) return;
    int d = t >> 2, h = t & 3;
    float s = 0.f;
    for (int c = 0; c < C_CH; ++c)
        s += We[d * HC + h * C_CH + c] * a_e[h * C_CH + c];
    Me[d * H_HEADS + h] = s;
}

// ---------------- per-edge logits + segment max (atomic) -------------------

__global__ void k_edge_alpha_max(const int* __restrict__ ei,
                                 const float* __restrict__ ef,
                                 const float* __restrict__ mean_ea,
                                 const float* __restrict__ Me,
                                 const float* __restrict__ as_,
                                 const float* __restrict__ ad_,
                                 float* __restrict__ alpha,
                                 float* __restrict__ m_) {
    int t = blockIdx.x * blockDim.x + threadIdx.x;
    if (t >= EP * H_HEADS) return;
    int e = t >> 2, h = t & 3;
    int s, d;
    const float* row;
    if (e < E_EDGES) { s = ei[e]; d = ei[E_EDGES + e]; row = ef + (size_t)e * EDGE_DIM; }
    else             { s = d = e - E_EDGES;            row = mean_ea + (size_t)s * EDGE_DIM; }
    float ae = 0.f;
#pragma unroll
    for (int k = 0; k < EDGE_DIM; ++k) ae += row[k] * Me[k * H_HEADS + h];
    float al = as_[s * H_HEADS + h] + ad_[d * H_HEADS + h] + ae;
    al = (al > 0.f) ? al : 0.2f * al;              // leaky_relu(0.2)
    alpha[t] = al;
    atomicMaxF(&m_[d * H_HEADS + h], al);
}

// ---------------- exp(alpha - max) + segment sum (atomic) ------------------

__global__ void k_edge_exp_sum(const int* __restrict__ ei,
                               float* __restrict__ alpha,
                               const float* __restrict__ m_,
                               float* __restrict__ den_) {
    int t = blockIdx.x * blockDim.x + threadIdx.x;
    if (t >= EP * H_HEADS) return;
    int e = t >> 2, h = t & 3;
    int d = (e < E_EDGES) ? ei[E_EDGES + e] : (e - E_EDGES);
    float ex = expf(alpha[t] - m_[d * H_HEADS + h]);
    alpha[t] = ex;
    atomicAdd(&den_[d * H_HEADS + h], ex);
}

// ---------- weighted scatter: agg[dst] += xp[src] * (ex/den[dst]) ----------

__global__ void k_scatter_agg(const int* __restrict__ ei,
                              const float* __restrict__ xp,
                              const float* __restrict__ ex,
                              const float* __restrict__ den_,
                              float* __restrict__ agg) {
    int e = blockIdx.x;
    int s, d;
    if (e < E_EDGES) { s = ei[e]; d = ei[E_EDGES + e]; }
    else             { s = d = e - E_EDGES; }
    const float* xr = xp + (size_t)s * HC;
    float*       ar = agg + (size_t)d * HC;
#pragma unroll
    for (int k = 0; k < 2; ++k) {
        int j = threadIdx.x + k * 256;
        int h = j >> 7;                       // j / C_CH
        float w = ex[e * H_HEADS + h] / den_[d * H_HEADS + h];
        atomicAdd(&ar[j], xr[j] * w);
    }
}

// ------------------------------ bias + ELU ---------------------------------

__global__ void k_bias_elu(float* __restrict__ z, const float* __restrict__ b) {
    int t = blockIdx.x * blockDim.x + threadIdx.x;
    if (t >= N_NODES * HC) return;
    float v = z[t] + b[t & (HC - 1)];
    z[t] = (v > 0.f) ? v : expm1f(v);
}

// -------- decoder: per edge sigmoid((u[src]+v[dst]+bd1) @ Wd2 + bd2) -------

__global__ void k_decoder(const int* __restrict__ ei,
                          const float* __restrict__ u,
                          const float* __restrict__ v,
                          const float* __restrict__ bd1,
                          const float* __restrict__ Wd2,
                          const float* __restrict__ bd2,
                          float* __restrict__ out) {
    int gid  = blockIdx.x * blockDim.x + threadIdx.x;
    int e    = gid >> 5;
    int lane = gid & 31;
    if (e >= E_EDGES) return;
    int s = ei[e], d = ei[E_EDGES + e];
    const float* up = u + (size_t)s * DEC_H;
    const float* vp = v + (size_t)d * DEC_H;
    float a0 = 0.f, a1 = 0.f, a2 = 0.f, a3 = 0.f;
#pragma unroll
    for (int k = 0; k < 8; ++k) {
        int j = lane * 8 + k;
        float hv = up[j] + vp[j] + bd1[j];
        const float* w = Wd2 + j * NUM_RELS;
        a0 += hv * w[0]; a1 += hv * w[1]; a2 += hv * w[2]; a3 += hv * w[3];
    }
    for (int off = 16; off > 0; off >>= 1) {
        a0 += __shfl_down(a0, off, 32);
        a1 += __shfl_down(a1, off, 32);
        a2 += __shfl_down(a2, off, 32);
        a3 += __shfl_down(a3, off, 32);
    }
    if (lane == 0) {
        out[e * 4 + 0] = 1.f / (1.f + expf(-(a0 + bd2[0])));
        out[e * 4 + 1] = 1.f / (1.f + expf(-(a1 + bd2[1])));
        out[e * 4 + 2] = 1.f / (1.f + expf(-(a2 + bd2[2])));
        out[e * 4 + 3] = 1.f / (1.f + expf(-(a3 + bd2[3])));
    }
}

// ------------------------------ host side ----------------------------------

static inline int cdiv(long long a, long long b) { return (int)((a + b - 1) / b); }

static void run_gemm(const unsigned short* A, const unsigned short* BT, float* C,
                     int M, int K, int Nc, hipStream_t s) {
    dim3 grid(M_PAD / 128, Nc / 64);
    k_gemm_bf16_wmma<<<grid, 256, 0, s>>>(A, BT, C, M, K, Nc);
}

struct Ws {
    float *bufA, *bufB, *mean, *cnt, *as_, *ad_, *m_, *den_, *alpha, *Me;
    unsigned* Ab;                 // bf16 activations [M_PAD, <=512], packed u32
    unsigned short *W1T, *W2T, *Wd1aT, *Wd1bT;   // bf16 transposed weights
};

static void run_layer(const unsigned short* Abf, int Fin,
                      const unsigned short* WT, const float* a_s, const float* a_d,
                      const float* We, const float* a_e, const float* b,
                      const int* ei, const float* ef,
                      const Ws& w, hipStream_t s) {
    // xp = A @ W -> bufA [N, 512]
    run_gemm(Abf, WT, w.bufA, N_NODES, Fin, HC, s);
    // per-node logits
    k_node_alpha<<<cdiv(N_NODES * H_HEADS, 256), 256, 0, s>>>(w.bufA, a_s, a_d, w.as_, w.ad_);
    // M_e = reduce(We, a_e)
    k_reduce_We_ae<<<1, 32, 0, s>>>(We, a_e, w.Me);
    // softmax stats init
    k_fill<<<cdiv(N_NODES * H_HEADS, 256), 256, 0, s>>>(w.m_, -INFINITY, N_NODES * H_HEADS);
    k_fill<<<cdiv(N_NODES * H_HEADS, 256), 256, 0, s>>>(w.den_, 0.f, N_NODES * H_HEADS);
    // edge logits + segment max
    k_edge_alpha_max<<<cdiv((long long)EP * H_HEADS, 256), 256, 0, s>>>(
        ei, ef, w.mean, w.Me, w.as_, w.ad_, w.alpha, w.m_);
    // exp + segment sum
    k_edge_exp_sum<<<cdiv((long long)EP * H_HEADS, 256), 256, 0, s>>>(ei, w.alpha, w.m_, w.den_);
    // aggregate into bufB
    k_fill<<<cdiv((long long)N_NODES * HC, 256), 256, 0, s>>>(w.bufB, 0.f, N_NODES * HC);
    k_scatter_agg<<<EP, 256, 0, s>>>(ei, w.bufA, w.alpha, w.den_, w.bufB);
    // bias + ELU in place -> z in bufB
    k_bias_elu<<<cdiv((long long)N_NODES * HC, 256), 256, 0, s>>>(w.bufB, b);
}

extern "C" void kernel_launch(void* const* d_in, const int* in_sizes, int n_in,
                              void* d_out, int out_size, void* d_ws, size_t ws_size,
                              hipStream_t stream) {
    const float* x   = (const float*)d_in[0];
    const int*   ei  = (const int*)  d_in[1];
    const float* ef  = (const float*)d_in[2];
    const float* W1  = (const float*)d_in[3];
    const float* as1 = (const float*)d_in[4];
    const float* ad1 = (const float*)d_in[5];
    const float* We1 = (const float*)d_in[6];
    const float* ae1 = (const float*)d_in[7];
    const float* b1  = (const float*)d_in[8];
    const float* W2  = (const float*)d_in[9];
    const float* as2 = (const float*)d_in[10];
    const float* ad2 = (const float*)d_in[11];
    const float* We2 = (const float*)d_in[12];
    const float* ae2 = (const float*)d_in[13];
    const float* b2  = (const float*)d_in[14];
    const float* Wd1 = (const float*)d_in[15];
    const float* bd1 = (const float*)d_in[16];
    const float* Wd2 = (const float*)d_in[17];
    const float* bd2 = (const float*)d_in[18];
    float* out = (float*)d_out;

    // ---- workspace layout (float units, everything 16B aligned) ----
    float* ws = (float*)d_ws;
    size_t off = 0;
    Ws w;
    w.bufA  = ws + off; off += (size_t)N_NODES * HC;        // xp / u|v
    w.bufB  = ws + off; off += (size_t)N_NODES * HC;        // agg / z
    w.mean  = ws + off; off += (size_t)N_NODES * EDGE_DIM;
    w.cnt   = ws + off; off += (size_t)N_NODES;
    w.as_   = ws + off; off += (size_t)N_NODES * H_HEADS;
    w.ad_   = ws + off; off += (size_t)N_NODES * H_HEADS;
    w.m_    = ws + off; off += (size_t)N_NODES * H_HEADS;
    w.den_  = ws + off; off += (size_t)N_NODES * H_HEADS;
    w.alpha = ws + off; off += (size_t)EP * H_HEADS;
    w.Me    = ws + off; off += 32;
    w.Ab    = (unsigned*)(ws + off); off += (size_t)M_PAD * HC / 2;   // bf16 [M_PAD,512]
    unsigned short* wT = (unsigned short*)(ws + off);
    w.W1T   = wT;                        // [512, 64]
    w.W2T   = wT + 512 * 64;             // [512, 512]
    w.Wd1aT = w.W2T + 512 * 512;         // [256, 512]
    w.Wd1bT = w.Wd1aT + 256 * 512;       // [256, 512]
    off += (512 * 64 + 512 * 512 + 2 * 256 * 512) / 2;
    (void)ws_size; (void)in_sizes; (void)n_in; (void)out_size;

    // ---- one-time weight conversion (bf16, transposed to [Nc,K]) ----
    k_cvt_transpose_bf16<<<cdiv(64 * 512, 256), 256, 0, stream>>>(W1, w.W1T, F_IN, HC);
    k_cvt_transpose_bf16<<<cdiv(512 * 512, 256), 256, 0, stream>>>(W2, w.W2T, HC, HC);
    k_cvt_transpose_bf16<<<cdiv(512 * 256, 256), 256, 0, stream>>>(Wd1, w.Wd1aT, HC, DEC_H);
    k_cvt_transpose_bf16<<<cdiv(512 * 256, 256), 256, 0, stream>>>(
        Wd1 + (size_t)HC * DEC_H, w.Wd1bT, HC, DEC_H);

    // ---- self-loop edge-attr mean (same for both layers) ----
    k_fill<<<cdiv(N_NODES * EDGE_DIM, 256), 256, 0, stream>>>(w.mean, 0.f, N_NODES * EDGE_DIM);
    k_fill<<<cdiv(N_NODES, 256), 256, 0, stream>>>(w.cnt, 0.f, N_NODES);
    k_edge_mean_accum<<<cdiv(E_EDGES, 256), 256, 0, stream>>>(ei, ef, w.mean, w.cnt);
    k_edge_mean_div<<<cdiv(N_NODES, 256), 256, 0, stream>>>(w.mean, w.cnt);

    // ---- layer 1: x -> bf16 padded, then GAT conv -> z1 in bufB ----
    k_cvt_bf16_pad<<<cdiv((long long)M_PAD * F_IN / 2, 256), 256, 0, stream>>>(
        x, w.Ab, N_NODES, F_IN, M_PAD);
    run_layer((const unsigned short*)w.Ab, F_IN, w.W1T, as1, ad1, We1, ae1, b1,
              ei, ef, w, stream);

    // ---- layer 2: z1 -> bf16 padded, GAT conv -> z2 in bufB ----
    k_cvt_bf16_pad<<<cdiv((long long)M_PAD * HC / 2, 256), 256, 0, stream>>>(
        w.bufB, w.Ab, N_NODES, HC, M_PAD);
    run_layer((const unsigned short*)w.Ab, HC, w.W2T, as2, ad2, We2, ae2, b2,
              ei, ef, w, stream);

    // ---- decoder: z2 -> bf16, u = z2@Wd1[:512], v = z2@Wd1[512:] ----
    k_cvt_bf16_pad<<<cdiv((long long)M_PAD * HC / 2, 256), 256, 0, stream>>>(
        w.bufB, w.Ab, N_NODES, HC, M_PAD);
    float* u = w.bufA;
    float* v = w.bufA + (size_t)N_NODES * DEC_H;
    run_gemm((const unsigned short*)w.Ab, w.Wd1aT, u, N_NODES, HC, DEC_H, stream);
    run_gemm((const unsigned short*)w.Ab, w.Wd1bT, v, N_NODES, HC, DEC_H, stream);

    // ---- per-edge: sigmoid((u[src]+v[dst]+bd1) @ Wd2 + bd2) ----
    k_decoder<<<cdiv((long long)E_EDGES * 32, 256), 256, 0, stream>>>(
        ei, u, v, bd1, Wd2, bd2, out);
}
